// Attention_63350767616384
// MI455X (gfx1250) — matrix operand order repaired and checked by reading, not compiled, not run on hardware
//
#include <hip/hip_runtime.h>

typedef __attribute__((ext_vector_type(16))) _Float16 v16h;
typedef __attribute__((ext_vector_type(8)))  _Float16 v8h;
typedef __attribute__((ext_vector_type(8)))  float    v8f;

#define L_SEQ 4096
#define D_MOD 512
#define NH    8
#define HD    64

union AFrag { v16h v; v8h h[2]; };

// ---- WMMA fragment loaders (CDNA5 ISA 7.12.2 layouts, wave32) ----

// A-matrix 16x32 f16: lanes 0-15 -> M=lane, lanes 16-31 -> M=lane-16.
// half i -> K = (i&7) + (i>=8 ? 16 : 0) + g*8   (g = lane>>4)
__device__ inline v16h load_a_frag(const _Float16* src, int ld, int row0, int col0) {
    int lane = threadIdx.x & 31;
    int g = lane >> 4, m = lane & 15;
    const _Float16* p = src + (size_t)(row0 + m) * ld + col0 + g * 8;
    AFrag a;
    a.h[0] = *(const v8h*)(p);
    a.h[1] = *(const v8h*)(p + 16);
    return a.v;
}

// B-matrix 32x16 f16: lane -> N = lane&15; half i -> K = (i&7) + (i>=8 ? 16 : 0) + g*8.
// Source is N-major (row = one output column, K contiguous): two 16B runs per lane.
__device__ inline v16h load_b_frag_nmajor(const _Float16* src, int ld, int n0, int col0) {
    int lane = threadIdx.x & 31;
    int g = lane >> 4, n = lane & 15;
    const _Float16* p = src + (size_t)(n0 + n) * ld + col0 + g * 8;
    AFrag b;
    b.h[0] = *(const v8h*)(p);
    b.h[1] = *(const v8h*)(p + 16);
    return b.v;
}

__device__ inline v8f wmma_f16(v16h a, v16h b, v8f c) {
    return __builtin_amdgcn_wmma_f32_16x16x32_f16(false, a, false, b, (short)0, c,
                                                  false, false);
}

// ---- CDNA5 async copy global -> LDS (ASYNCcnt-tracked) ----
__device__ inline uint32_t lds_off32(const void* p) {
    // flat LDS aperture: addr[31:0] is the wave-relative LDS byte address
    return (uint32_t)(uintptr_t)p;
}
__device__ inline void async_copy_b128(uint32_t lds_byte, const void* gaddr) {
    asm volatile("global_load_async_to_lds_b128 %0, %1, off"
                 :: "v"(lds_byte), "v"((uint64_t)(uintptr_t)gaddr)
                 : "memory");
}
__device__ inline void wait_async0() {
    asm volatile("s_wait_asynccnt 0" ::: "memory");
}

// ---- prep kernels ----

__global__ void add_pe_f16(const float* __restrict__ x, const float* __restrict__ pe,
                           _Float16* __restrict__ xp, int n) {
    int i = blockIdx.x * blockDim.x + threadIdx.x;
    if (i < n) xp[i] = (_Float16)(x[i] + pe[i]);
}

// w: (rows x cols) f32 row-major -> wt: (cols x rows) f16 (N-major for B frags)
__global__ void transpose_to_f16(const float* __restrict__ w, _Float16* __restrict__ wt,
                                 int rows, int cols) {
    int i = blockIdx.x * blockDim.x + threadIdx.x;
    if (i >= rows * cols) return;
    int n = i / rows;
    int k = i - n * rows;
    wt[i] = (_Float16)w[(size_t)k * cols + n];
}

// ---- fused QKV projection: [L x D] @ [D x D] per matrix ----
// block = 4 waves; wave: 32 rows x 64 cols (8 C tiles, 2 A frags share 4 B frags).
__global__ __launch_bounds__(128)
void qkv_gemm(const _Float16* __restrict__ xp,
              const _Float16* __restrict__ wqT, const _Float16* __restrict__ wkT,
              const _Float16* __restrict__ wvT,
              const float* __restrict__ bq, const float* __restrict__ bk,
              const float* __restrict__ bv,
              _Float16* __restrict__ qb, _Float16* __restrict__ kb,
              _Float16* __restrict__ vT) {
    int which = blockIdx.z;               // 0=Q, 1=K, 2=V
    int h = blockIdx.y;
    int row0 = blockIdx.x * 128;
    int w = threadIdx.x >> 5;
    int lane = threadIdx.x & 31;
    int g = lane >> 4, lc = lane & 15;
    int mrow = row0 + w * 32;

    const _Float16* wT = (which == 0) ? wqT : (which == 1) ? wkT : wvT;
    const float*    bi = (which == 0) ? bq  : (which == 1) ? bk  : bv;

    v8f acc[2][4] = {};
    for (int kk = 0; kk < D_MOD; kk += 32) {
        v16h a0 = load_a_frag(xp, D_MOD, mrow, kk);
        v16h a1 = load_a_frag(xp, D_MOD, mrow + 16, kk);
#pragma unroll
        for (int nt = 0; nt < 4; ++nt) {
            v16h b = load_b_frag_nmajor(wT, D_MOD, h * 64 + nt * 16, kk);
            acc[0][nt] = wmma_f16(a0, b, acc[0][nt]);
            acc[1][nt] = wmma_f16(a1, b, acc[1][nt]);
        }
    }
#pragma unroll
    for (int ph = 0; ph < 2; ++ph) {
#pragma unroll
        for (int nt = 0; nt < 4; ++nt) {
            int d = nt * 16 + lc;
            float bval = bi[h * HD + d];
#pragma unroll
            for (int r = 0; r < 8; ++r) {
                int m = mrow + ph * 16 + r + 8 * g;
                float val = acc[ph][nt][r] + bval;
                if (which == 0)
                    qb[((size_t)h * L_SEQ + m) * HD + d] = (_Float16)(val * 0.125f);
                else if (which == 1)
                    kb[((size_t)h * L_SEQ + m) * HD + d] = (_Float16)val;
                else
                    vT[((size_t)h * HD + d) * L_SEQ + m] = (_Float16)val;   // transposed
            }
        }
    }
}

// ---- flash attention with relative bias ----
// grid (L/64, H); block = 4 waves; wave owns 16 q-rows.
// K/V 64-key tiles staged once per block via async copy, double-buffered.
__global__ __launch_bounds__(128)
void flash_attn(const _Float16* __restrict__ qb, const _Float16* __restrict__ kb,
                const _Float16* __restrict__ vT, const float* __restrict__ rel_bias,
                _Float16* __restrict__ ob) {
    __shared__ __align__(16) _Float16 ktile[2][64 * 64];  // [key][d]
    __shared__ __align__(16) _Float16 vtile[2][64 * 64];  // [d][key]
    __shared__ __align__(16) _Float16 lds_p[4][16 * 64];

    int h = blockIdx.y;
    int q0 = blockIdx.x * 64;
    int tid = threadIdx.x;
    int w = tid >> 5;
    int lane = tid & 31;
    int g = lane >> 4, lc = lane & 15;
    int qw = q0 + w * 16;

    const _Float16* qh = qb + (size_t)h * L_SEQ * HD;
    const _Float16* kh = kb + (size_t)h * L_SEQ * HD;
    const _Float16* vh = vT + (size_t)h * HD * L_SEQ;
    const float*   rbh = rel_bias + (size_t)h * (2 * L_SEQ);

    v16h aq0 = load_a_frag(qh, HD, qw, 0);
    v16h aq1 = load_a_frag(qh, HD, qw, 32);

    float mrun[8], lrun[8];
#pragma unroll
    for (int r = 0; r < 8; ++r) { mrun[r] = -1e30f; lrun[r] = 0.0f; }
    v8f o[4] = {};

    _Float16* myp = lds_p[w];

    // cooperative async stage of one 64-key tile (K: contiguous 8KB; V: 64 x 128B rows)
    auto issue_tile = [&](int buf, int kbase) {
        const char* kg = (const char*)(kh + (size_t)kbase * HD);
        uint32_t klds = lds_off32(&ktile[buf][0]);
#pragma unroll
        for (int c = 0; c < 4; ++c) {
            int idx = tid + c * 128;                 // 512 x 16B chunks
            async_copy_b128(klds + idx * 16, kg + idx * 16);
        }
        const _Float16* vg = vh + kbase;
        uint32_t vlds = lds_off32(&vtile[buf][0]);
#pragma unroll
        for (int c = 0; c < 4; ++c) {
            int idx = tid + c * 128;                 // row = idx/8 (d), chunk = idx%8
            int row = idx >> 3, ch = idx & 7;
            async_copy_b128(vlds + row * 128 + ch * 16,
                            (const char*)(vg + (size_t)row * L_SEQ) + ch * 16);
        }
    };

    issue_tile(0, 0);

    for (int kt = 0; kt < L_SEQ / 64; ++kt) {
        int kbase = kt * 64;
        int cur = kt & 1;
        wait_async0();
        __syncthreads();                 // tile[cur] visible to all waves
        if (kt + 1 < L_SEQ / 64)
            issue_tile(cur ^ 1, kbase + 64);

        const _Float16* kbuf = &ktile[cur][0];
        const _Float16* vbuf = &vtile[cur][0];

        // S = Q @ K^T  (16x64 per wave), K fragments from LDS
        v8f s[4] = {};
#pragma unroll
        for (int nt = 0; nt < 4; ++nt) {
            v16h b0 = load_b_frag_nmajor(kbuf, 64, nt * 16, 0);
            v16h b1 = load_b_frag_nmajor(kbuf, 64, nt * 16, 32);
            s[nt] = wmma_f16(aq0, b0, s[nt]);
            s[nt] = wmma_f16(aq1, b1, s[nt]);
        }
        // relative bias: bias[h,q,k] = rel_bias[h, k - q + L - 1]
#pragma unroll
        for (int nt = 0; nt < 4; ++nt) {
            int n = kbase + nt * 16 + lc;
#pragma unroll
            for (int r = 0; r < 8; ++r) {
                int m = qw + r + 8 * g;
                s[nt][r] += rbh[n - m + L_SEQ - 1];
            }
        }
        // online softmax: row lives across 16 lanes (same half) per C VGPR
        float mnew[8], scale[8];
#pragma unroll
        for (int r = 0; r < 8; ++r) {
            float v = fmaxf(fmaxf(s[0][r], s[1][r]), fmaxf(s[2][r], s[3][r]));
#pragma unroll
            for (int off = 1; off < 16; off <<= 1)
                v = fmaxf(v, __shfl_xor(v, off, 32));
            mnew[r]  = fmaxf(mrun[r], v);
            scale[r] = __expf(mrun[r] - mnew[r]);
            mrun[r]  = mnew[r];
        }
        // P = exp(S - mnew); rowsum; stage P in LDS (C-layout -> A-layout bridge)
#pragma unroll
        for (int r = 0; r < 8; ++r) {
            float rsum = 0.0f;
#pragma unroll
            for (int nt = 0; nt < 4; ++nt) {
                float p = __expf(s[nt][r] - mnew[r]);
                rsum += p;
                myp[(r + 8 * g) * 64 + nt * 16 + lc] = (_Float16)p;
            }
#pragma unroll
            for (int off = 1; off < 16; off <<= 1)
                rsum += __shfl_xor(rsum, off, 32);
            lrun[r] = lrun[r] * scale[r] + rsum;
        }
#pragma unroll
        for (int dt = 0; dt < 4; ++dt)
#pragma unroll
            for (int r = 0; r < 8; ++r)
                o[dt][r] *= scale[r];
        // per-wave LDS store->load ordering for the P staging buffer
        asm volatile("s_wait_dscnt 0" ::: "memory");
        v16h ap0 = load_a_frag(myp, 64, 0, 0);
        v16h ap1 = load_a_frag(myp, 64, 0, 32);
        // O += P @ V, V fragments from LDS
#pragma unroll
        for (int dt = 0; dt < 4; ++dt) {
            v16h bv0 = load_b_frag_nmajor(vbuf, 64, dt * 16, 0);
            v16h bv1 = load_b_frag_nmajor(vbuf, 64, dt * 16, 32);
            o[dt] = wmma_f16(ap0, bv0, o[dt]);
            o[dt] = wmma_f16(ap1, bv1, o[dt]);
        }
        __syncthreads();                 // all waves done with tile[cur] before reuse
    }
    // normalize, store O as (L x 512) f16 for the output projection
#pragma unroll
    for (int r = 0; r < 8; ++r) lrun[r] = 1.0f / lrun[r];
#pragma unroll
    for (int dt = 0; dt < 4; ++dt) {
        int d = dt * 16 + lc;
#pragma unroll
        for (int r = 0; r < 8; ++r) {
            int m = qw + r + 8 * g;
            ob[(size_t)m * D_MOD + h * HD + d] = (_Float16)(o[dt][r] * lrun[r]);
        }
    }
}

// ---- output projection: [L x 512] @ [512 x 512] + bo -> f32 ----
__global__ __launch_bounds__(128)
void out_proj(const _Float16* __restrict__ ob, const _Float16* __restrict__ woT,
              const float* __restrict__ bo, float* __restrict__ out) {
    int nblk = blockIdx.y;
    int row0 = blockIdx.x * 128;
    int w = threadIdx.x >> 5;
    int lane = threadIdx.x & 31;
    int g = lane >> 4, lc = lane & 15;
    int mrow = row0 + w * 32;

    v8f acc[2][4] = {};
    for (int kk = 0; kk < D_MOD; kk += 32) {
        v16h a0 = load_a_frag(ob, D_MOD, mrow, kk);
        v16h a1 = load_a_frag(ob, D_MOD, mrow + 16, kk);
#pragma unroll
        for (int nt = 0; nt < 4; ++nt) {
            v16h b = load_b_frag_nmajor(woT, D_MOD, nblk * 64 + nt * 16, kk);
            acc[0][nt] = wmma_f16(a0, b, acc[0][nt]);
            acc[1][nt] = wmma_f16(a1, b, acc[1][nt]);
        }
    }
#pragma unroll
    for (int ph = 0; ph < 2; ++ph) {
#pragma unroll
        for (int nt = 0; nt < 4; ++nt) {
            int n = nblk * 64 + nt * 16 + lc;
            float b = bo[n];
#pragma unroll
            for (int r = 0; r < 8; ++r) {
                int m = mrow + ph * 16 + r + 8 * g;
                out[(size_t)m * D_MOD + n] = acc[ph][nt][r] + b;
            }
        }
    }
}

extern "C" void kernel_launch(void* const* d_in, const int* in_sizes, int n_in,
                              void* d_out, int out_size, void* d_ws, size_t ws_size,
                              hipStream_t stream) {
    (void)in_sizes; (void)n_in; (void)out_size; (void)ws_size;
    const float* x  = (const float*)d_in[0];
    const float* pe = (const float*)d_in[1];
    const float* rb = (const float*)d_in[2];
    const float* Wq = (const float*)d_in[3];
    const float* bq = (const float*)d_in[4];
    const float* Wk = (const float*)d_in[5];
    const float* bk = (const float*)d_in[6];
    const float* Wv = (const float*)d_in[7];
    const float* bv = (const float*)d_in[8];
    const float* Wo = (const float*)d_in[9];
    const float* bo = (const float*)d_in[10];
    float* out = (float*)d_out;

    char* ws = (char*)d_ws;
    size_t off = 0;
    auto alloc = [&](size_t bytes) {
        char* p = ws + off;
        off += (bytes + 255) & ~(size_t)255;
        return p;
    };
    _Float16* xp  = (_Float16*)alloc(sizeof(_Float16) * L_SEQ * D_MOD);
    _Float16* wqT = (_Float16*)alloc(sizeof(_Float16) * D_MOD * D_MOD);
    _Float16* wkT = (_Float16*)alloc(sizeof(_Float16) * D_MOD * D_MOD);
    _Float16* wvT = (_Float16*)alloc(sizeof(_Float16) * D_MOD * D_MOD);
    _Float16* woT = (_Float16*)alloc(sizeof(_Float16) * D_MOD * D_MOD);
    _Float16* qb  = (_Float16*)alloc(sizeof(_Float16) * NH * L_SEQ * HD);
    _Float16* kb  = (_Float16*)alloc(sizeof(_Float16) * NH * L_SEQ * HD);
    _Float16* vT  = (_Float16*)alloc(sizeof(_Float16) * NH * HD * L_SEQ);
    _Float16* ob  = (_Float16*)alloc(sizeof(_Float16) * L_SEQ * D_MOD);

    int nxp = L_SEQ * D_MOD;
    add_pe_f16<<<(nxp + 255) / 256, 256, 0, stream>>>(x, pe, xp, nxp);

    int nw = D_MOD * D_MOD;
    transpose_to_f16<<<(nw + 255) / 256, 256, 0, stream>>>(Wq, wqT, D_MOD, D_MOD);
    transpose_to_f16<<<(nw + 255) / 256, 256, 0, stream>>>(Wk, wkT, D_MOD, D_MOD);
    transpose_to_f16<<<(nw + 255) / 256, 256, 0, stream>>>(Wv, wvT, D_MOD, D_MOD);
    transpose_to_f16<<<(nw + 255) / 256, 256, 0, stream>>>(Wo, woT, D_MOD, D_MOD);

    qkv_gemm<<<dim3(L_SEQ / 128, NH, 3), 128, 0, stream>>>(xp, wqT, wkT, wvT,
                                                           bq, bk, bv, qb, kb, vT);

    flash_attn<<<dim3(L_SEQ / 64, NH), 128, 0, stream>>>(qb, kb, vT, rb, ob);

    out_proj<<<dim3(L_SEQ / 128, NH), 128, 0, stream>>>(ob, woT, bo, out);
}
